// RoPEAttention_18811956757041
// MI455X (gfx1250) — matrix operand order repaired and checked by reading, not compiled, not run on hardware
//
#include <hip/hip_runtime.h>
#include <stdint.h>

typedef __attribute__((ext_vector_type(16))) __bf16 v16bf;
typedef __attribute__((ext_vector_type(8)))  float  v8f;
typedef unsigned short ushort_t;

#define S_LEN   2048
#define D_MOD   1024
#define N_HEADS 16
#define HEAD_D  64
#define B_SZ    2
#define M_ROWS  (B_SZ * S_LEN)   // 4096

union FragU { uint4 q[2]; v16bf v; };

__device__ __forceinline__ ushort_t f2bf(float f) {
    unsigned u = __float_as_uint(f);
    u += 0x7FFFu + ((u >> 16) & 1u);   // round-to-nearest-even
    return (ushort_t)(u >> 16);
}

// A-matrix 16x32 bf16 fragment (ISA 7.12.2): lane row = lane&15;
// elems 0..7 -> K = kbase + hi*8 .. ; elems 8..15 -> K = kbase+16+hi*8 ..
__device__ __forceinline__ v16bf load_a_frag(const ushort_t* row, int kbase, int hi) {
    FragU f;
    f.q[0] = *(const uint4*)(row + kbase + hi * 8);
    f.q[1] = *(const uint4*)(row + kbase + 16 + hi * 8);
    return f.v;
}

// B-matrix 32x16 bf16 fragment: lane col = lane&15; same K order as A:
// elems 0..7 -> K = kbase + hi*8 .. ; elems 8..15 -> K = kbase+16+hi*8 ..
__device__ __forceinline__ v16bf load_b_frag(const ushort_t* row, int kbase, int hi) {
    FragU f;
    f.q[0] = *(const uint4*)(row + kbase + hi * 8);
    f.q[1] = *(const uint4*)(row + kbase + 16 + hi * 8);
    return f.v;
}

#define WMMA_BF16(a, b, c) \
    __builtin_amdgcn_wmma_f32_16x16x32_bf16(false, (a), false, (b), (short)0, (c), false, false)

// ---------------------------------------------------------------------------
// fp32 -> bf16, 4 elements per thread (vectorized)
__global__ void k_f32_to_bf16(const float* __restrict__ src, ushort_t* __restrict__ dst, int n4) {
    int i = blockIdx.x * blockDim.x + threadIdx.x;
    if (i < n4) {
        float4 v = ((const float4*)src)[i];
        ushort_t r0 = f2bf(v.x), r1 = f2bf(v.y), r2 = f2bf(v.z), r3 = f2bf(v.w);
        uint2 o;
        o.x = (unsigned)r0 | ((unsigned)r1 << 16);
        o.y = (unsigned)r2 | ((unsigned)r3 << 16);
        ((uint2*)dst)[i] = o;
    }
}

// ---------------------------------------------------------------------------
// C = X(bf16 [4096,1024]) @ W^T(bf16 [1024,1024]); one wave -> 16 rows x 64 cols
// (one full head slice). Software-pipelined: k+32 fragments loaded under the
// k-step WMMAs. mode 0: RoPE -> [B,H,S,hd].  mode 2: store V^T [B,H,hd,S].
__global__ void __launch_bounds__(256)
k_qkv_gemm(const ushort_t* __restrict__ X, const ushort_t* __restrict__ W,
           ushort_t* __restrict__ dst, int mode) {
    const int tid  = threadIdx.x;
    const int lane = tid & 31;
    const int wv   = tid >> 5;
    const int ln   = lane & 15;
    const int hi   = lane >> 4;
    const int gw   = blockIdx.x * 8 + wv;   // 0..4095
    const int mt   = gw >> 4;               // row tile 0..255
    const int n64  = gw & 15;               // head index
    const int mbase = mt * 16;

    const ushort_t* arow = X + (size_t)(mbase + ln) * D_MOD;
    const ushort_t* br0  = W + (size_t)(n64 * 64 +  0 + ln) * D_MOD;
    const ushort_t* br1  = W + (size_t)(n64 * 64 + 16 + ln) * D_MOD;
    const ushort_t* br2  = W + (size_t)(n64 * 64 + 32 + ln) * D_MOD;
    const ushort_t* br3  = W + (size_t)(n64 * 64 + 48 + ln) * D_MOD;

    v8f acc0 = {0,0,0,0,0,0,0,0};
    v8f acc1 = acc0, acc2 = acc0, acc3 = acc0;

    // prologue: fragments for k = 0
    v16bf a  = load_a_frag(arow, 0, hi);
    v16bf b0 = load_b_frag(br0, 0, hi);
    v16bf b1 = load_b_frag(br1, 0, hi);
    v16bf b2 = load_b_frag(br2, 0, hi);
    v16bf b3 = load_b_frag(br3, 0, hi);

    #pragma unroll 4
    for (int k = 0; k < D_MOD; k += 32) {
        const int kn = (k + 32 < D_MOD) ? (k + 32) : 0;   // tail wraps (discarded)
        v16bf an  = load_a_frag(arow, kn, hi);
        v16bf nb0 = load_b_frag(br0, kn, hi);
        v16bf nb1 = load_b_frag(br1, kn, hi);
        v16bf nb2 = load_b_frag(br2, kn, hi);
        v16bf nb3 = load_b_frag(br3, kn, hi);
        acc0 = WMMA_BF16(a, b0, acc0);
        acc1 = WMMA_BF16(a, b1, acc1);
        acc2 = WMMA_BF16(a, b2, acc2);
        acc3 = WMMA_BF16(a, b3, acc3);
        a = an; b0 = nb0; b1 = nb1; b2 = nb2; b3 = nb3;
    }

    const int h = n64;
    if (mode == 2) {
        // V transposed: dst[((b*H+h)*hd + d) * S + s]
        #pragma unroll
        for (int e = 0; e < 8; ++e) {
            int m = mbase + e + 8 * hi;
            int s = m & (S_LEN - 1);
            int b = m >> 11;
            size_t base = (size_t)(b * N_HEADS + h) * HEAD_D;
            dst[(base +  0 + ln) * S_LEN + s] = f2bf(acc0[e]);
            dst[(base + 16 + ln) * S_LEN + s] = f2bf(acc1[e]);
            dst[(base + 32 + ln) * S_LEN + s] = f2bf(acc2[e]);
            dst[(base + 48 + ln) * S_LEN + s] = f2bf(acc3[e]);
        }
    } else {
        // RoPE fused epilogue: pairs (d, d+32) are acc tiles (t, t^2), same lane/elem.
        const float kc = 9.210340371976184f / 32.0f;   // ln(10000)/32
        float invf0 = __expf(-(float)(ln)      * kc);  // d%32 == ln      (tiles 0,2)
        float invf1 = __expf(-(float)(ln + 16) * kc);  // d%32 == 16+ln   (tiles 1,3)
        #pragma unroll
        for (int e = 0; e < 8; ++e) {
            int m = mbase + e + 8 * hi;
            int s = m & (S_LEN - 1);
            int b = m >> 11;
            float fs = (float)s;
            float a0 = fs * invf0, a1 = fs * invf1;
            float c0 = __cosf(a0), s0 = __sinf(a0);
            float c1 = __cosf(a1), s1 = __sinf(a1);
            float o0 = acc0[e] * c0 - acc2[e] * s0;
            float o1 = acc1[e] * c1 - acc3[e] * s1;
            float o2 = acc2[e] * c0 + acc0[e] * s0;
            float o3 = acc3[e] * c1 + acc1[e] * s1;
            size_t rb = ((size_t)(b * N_HEADS + h) * S_LEN + s) * HEAD_D;
            dst[rb +  0 + ln] = f2bf(o0);
            dst[rb + 16 + ln] = f2bf(o1);
            dst[rb + 32 + ln] = f2bf(o2);
            dst[rb + 48 + ln] = f2bf(o3);
        }
    }
}

// ---------------------------------------------------------------------------
// Flash-style attention: one wave = 16 query rows of one (b,h).
// 64 key-blocks of 32: 4 score WMMAs, online softmax (shfl_xor within 16-lane
// halves == rows of the C layout), P relaid C->A layout via per-wave LDS,
// 4 PV WMMAs. K fragments double-buffered across blocks; V fragments issued
// before the softmax VALU section so their latency hides under it.
__global__ void __launch_bounds__(256)
k_attention(const ushort_t* __restrict__ Q, const ushort_t* __restrict__ K,
            const ushort_t* __restrict__ Vt, ushort_t* __restrict__ O) {
    __shared__ ushort_t ldsP[8][16][32];
    const int tid  = threadIdx.x;
    const int lane = tid & 31;
    const int wv   = tid >> 5;
    const int ln   = lane & 15;
    const int hi   = lane >> 4;
    const int gw   = blockIdx.x * 8 + wv;   // 0..4095
    const int bh   = gw >> 7;               // 0..31
    const int mt   = gw & 127;
    const int sbase = mt * 16;
    const int b = bh >> 4, h = bh & 15;

    const ushort_t* Qbh = Q  + (size_t)bh * S_LEN * HEAD_D;
    const ushort_t* Kbh = K  + (size_t)bh * S_LEN * HEAD_D;
    const ushort_t* Vbh = Vt + (size_t)bh * HEAD_D * S_LEN;

    const ushort_t* qrow = Qbh + (size_t)(sbase + ln) * HEAD_D;
    v16bf aq0 = load_a_frag(qrow, 0,  hi);
    v16bf aq1 = load_a_frag(qrow, 32, hi);

    const ushort_t* vt0 = Vbh + (size_t)( 0 + ln) * S_LEN;
    const ushort_t* vt1 = Vbh + (size_t)(16 + ln) * S_LEN;
    const ushort_t* vt2 = Vbh + (size_t)(32 + ln) * S_LEN;
    const ushort_t* vt3 = Vbh + (size_t)(48 + ln) * S_LEN;

    v8f zf = {0,0,0,0,0,0,0,0};
    v8f o0 = zf, o1 = zf, o2 = zf, o3 = zf;
    float mrun[8], lrun[8];
    #pragma unroll
    for (int e = 0; e < 8; ++e) { mrun[e] = -1e30f; lrun[e] = 0.0f; }

    ushort_t* myP = &ldsP[wv][0][0];

    // prologue: K fragments for key-block 0
    v16bf bk0l = load_b_frag(Kbh + (size_t)( 0 + ln) * HEAD_D, 0,  hi);
    v16bf bk0h = load_b_frag(Kbh + (size_t)( 0 + ln) * HEAD_D, 32, hi);
    v16bf bk1l = load_b_frag(Kbh + (size_t)(16 + ln) * HEAD_D, 0,  hi);
    v16bf bk1h = load_b_frag(Kbh + (size_t)(16 + ln) * HEAD_D, 32, hi);

    for (int kb = 0; kb < S_LEN; kb += 32) {
        // scores for current block
        v8f c0 = zf, c1 = zf;
        c0 = WMMA_BF16(aq0, bk0l, c0);
        c0 = WMMA_BF16(aq1, bk0h, c0);
        c1 = WMMA_BF16(aq0, bk1l, c1);
        c1 = WMMA_BF16(aq1, bk1h, c1);

        // issue next block's K fragments (tail wraps; values discarded)
        const int kn = (kb + 32) & (S_LEN - 1);
        const ushort_t* nr0 = Kbh + (size_t)(kn +      ln) * HEAD_D;
        const ushort_t* nr1 = Kbh + (size_t)(kn + 16 + ln) * HEAD_D;
        v16bf tk0l = load_b_frag(nr0, 0,  hi);
        v16bf tk0h = load_b_frag(nr0, 32, hi);
        v16bf tk1l = load_b_frag(nr1, 0,  hi);
        v16bf tk1h = load_b_frag(nr1, 32, hi);

        // issue this block's V fragments; latency hidden by softmax VALU below
        v16bf bv0 = load_b_frag(vt0, kb, hi);
        v16bf bv1 = load_b_frag(vt1, kb, hi);
        v16bf bv2 = load_b_frag(vt2, kb, hi);
        v16bf bv3 = load_b_frag(vt3, kb, hi);

        float p0[8], p1[8];
        #pragma unroll
        for (int e = 0; e < 8; ++e) {
            float x0 = c0[e] * 0.125f;        // 1/sqrt(64)
            float x1 = c1[e] * 0.125f;
            float bm = fmaxf(x0, x1);
            bm = fmaxf(bm, __shfl_xor(bm, 1, 32));
            bm = fmaxf(bm, __shfl_xor(bm, 2, 32));
            bm = fmaxf(bm, __shfl_xor(bm, 4, 32));
            bm = fmaxf(bm, __shfl_xor(bm, 8, 32));
            float mnew  = fmaxf(mrun[e], bm);
            float alpha = __expf(mrun[e] - mnew);
            p0[e] = __expf(x0 - mnew);
            p1[e] = __expf(x1 - mnew);
            float bs = p0[e] + p1[e];
            bs += __shfl_xor(bs, 1, 32);
            bs += __shfl_xor(bs, 2, 32);
            bs += __shfl_xor(bs, 4, 32);
            bs += __shfl_xor(bs, 8, 32);
            lrun[e] = lrun[e] * alpha + bs;
            mrun[e] = mnew;
            o0[e] *= alpha; o1[e] *= alpha; o2[e] *= alpha; o3[e] *= alpha;
        }

        // P: C-layout -> row-major 16x32 bf16 tile in per-wave LDS
        #pragma unroll
        for (int e = 0; e < 8; ++e) {
            int r = e + 8 * hi;
            myP[r * 32 + ln]      = f2bf(p0[e]);
            myP[r * 32 + 16 + ln] = f2bf(p1[e]);
        }
        asm volatile("s_wait_dscnt 0" ::: "memory");   // DS ops in-order per wave
        v16bf pa = load_a_frag(myP + (size_t)ln * 32, 0, hi);

        o0 = WMMA_BF16(pa, bv0, o0);
        o1 = WMMA_BF16(pa, bv1, o1);
        o2 = WMMA_BF16(pa, bv2, o2);
        o3 = WMMA_BF16(pa, bv3, o3);

        bk0l = tk0l; bk0h = tk0h; bk1l = tk1l; bk1h = tk1h;
    }

    #pragma unroll
    for (int e = 0; e < 8; ++e) {
        float inv = 1.0f / lrun[e];
        int s = sbase + e + 8 * hi;
        size_t rb = ((size_t)(b * S_LEN) + s) * D_MOD + h * HEAD_D;
        O[rb +  0 + ln] = f2bf(o0[e] * inv);
        O[rb + 16 + ln] = f2bf(o1[e] * inv);
        O[rb + 32 + ln] = f2bf(o2[e] * inv);
        O[rb + 48 + ln] = f2bf(o3[e] * inv);
    }
}

// ---------------------------------------------------------------------------
// out(f32) = Attn(bf16 [4096,1024]) @ Wo^T(bf16 [1024,1024]); same pipelining.
__global__ void __launch_bounds__(256)
k_out_gemm(const ushort_t* __restrict__ X, const ushort_t* __restrict__ W,
           float* __restrict__ out) {
    const int tid  = threadIdx.x;
    const int lane = tid & 31;
    const int wv   = tid >> 5;
    const int ln   = lane & 15;
    const int hi   = lane >> 4;
    const int gw   = blockIdx.x * 8 + wv;
    const int mt   = gw >> 4;
    const int n64  = gw & 15;
    const int mbase = mt * 16;

    const ushort_t* arow = X + (size_t)(mbase + ln) * D_MOD;
    const ushort_t* br0  = W + (size_t)(n64 * 64 +  0 + ln) * D_MOD;
    const ushort_t* br1  = W + (size_t)(n64 * 64 + 16 + ln) * D_MOD;
    const ushort_t* br2  = W + (size_t)(n64 * 64 + 32 + ln) * D_MOD;
    const ushort_t* br3  = W + (size_t)(n64 * 64 + 48 + ln) * D_MOD;

    v8f acc0 = {0,0,0,0,0,0,0,0};
    v8f acc1 = acc0, acc2 = acc0, acc3 = acc0;

    v16bf a  = load_a_frag(arow, 0, hi);
    v16bf b0 = load_b_frag(br0, 0, hi);
    v16bf b1 = load_b_frag(br1, 0, hi);
    v16bf b2 = load_b_frag(br2, 0, hi);
    v16bf b3 = load_b_frag(br3, 0, hi);

    #pragma unroll 4
    for (int k = 0; k < D_MOD; k += 32) {
        const int kn = (k + 32 < D_MOD) ? (k + 32) : 0;
        v16bf an  = load_a_frag(arow, kn, hi);
        v16bf nb0 = load_b_frag(br0, kn, hi);
        v16bf nb1 = load_b_frag(br1, kn, hi);
        v16bf nb2 = load_b_frag(br2, kn, hi);
        v16bf nb3 = load_b_frag(br3, kn, hi);
        acc0 = WMMA_BF16(a, b0, acc0);
        acc1 = WMMA_BF16(a, b1, acc1);
        acc2 = WMMA_BF16(a, b2, acc2);
        acc3 = WMMA_BF16(a, b3, acc3);
        a = an; b0 = nb0; b1 = nb1; b2 = nb2; b3 = nb3;
    }

    #pragma unroll
    for (int e = 0; e < 8; ++e) {
        size_t rb = (size_t)(mbase + e + 8 * hi) * D_MOD + n64 * 64;
        out[rb +  0 + ln] = acc0[e];
        out[rb + 16 + ln] = acc1[e];
        out[rb + 32 + ln] = acc2[e];
        out[rb + 48 + ln] = acc3[e];
    }
}

// ---------------------------------------------------------------------------
extern "C" void kernel_launch(void* const* d_in, const int* in_sizes, int n_in,
                              void* d_out, int out_size, void* d_ws, size_t ws_size,
                              hipStream_t stream) {
    const float* hs = (const float*)d_in[0];
    const float* wq = (const float*)d_in[1];
    const float* wk = (const float*)d_in[2];
    const float* wv = (const float*)d_in[3];
    const float* wo = (const float*)d_in[4];
    float* out = (float*)d_out;

    const size_t NHS = (size_t)M_ROWS * D_MOD;                        // 4,194,304
    const size_t NW  = (size_t)D_MOD * D_MOD;                         // 1,048,576
    const size_t NQ  = (size_t)B_SZ * N_HEADS * S_LEN * HEAD_D;       // 4,194,304

    ushort_t* p   = (ushort_t*)d_ws;
    ushort_t* hsb = p; p += NHS;
    ushort_t* wqb = p; p += NW;
    ushort_t* wkb = p; p += NW;
    ushort_t* wvb = p; p += NW;
    ushort_t* wob = p; p += NW;
    ushort_t* Qb  = p; p += NQ;
    ushort_t* Kb  = p; p += NQ;
    ushort_t* Vtb = p; p += NQ;
    ushort_t* Ob  = p; p += NHS;

    k_f32_to_bf16<<<(int)(NHS / 1024), 256, 0, stream>>>(hs, hsb, (int)(NHS / 4));
    k_f32_to_bf16<<<(int)(NW  / 1024), 256, 0, stream>>>(wq, wqb, (int)(NW / 4));
    k_f32_to_bf16<<<(int)(NW  / 1024), 256, 0, stream>>>(wk, wkb, (int)(NW / 4));
    k_f32_to_bf16<<<(int)(NW  / 1024), 256, 0, stream>>>(wv, wvb, (int)(NW / 4));
    k_f32_to_bf16<<<(int)(NW  / 1024), 256, 0, stream>>>(wo, wob, (int)(NW / 4));

    k_qkv_gemm<<<512, 256, 0, stream>>>(hsb, wqb, Qb,  0);   // Q + RoPE
    k_qkv_gemm<<<512, 256, 0, stream>>>(hsb, wkb, Kb,  0);   // K + RoPE
    k_qkv_gemm<<<512, 256, 0, stream>>>(hsb, wvb, Vtb, 2);   // V transposed

    k_attention<<<512, 256, 0, stream>>>(Qb, Kb, Vtb, Ob);

    k_out_gemm<<<512, 256, 0, stream>>>(Ob, wob, out);
}
